// MPNEncoder_83743272337589
// MI455X (gfx1250) — matrix operand order repaired and checked by reading, not compiled, not run on hardware
//
#include <hip/hip_runtime.h>
#include <hip/hip_bf16.h>

typedef __attribute__((ext_vector_type(16))) _Float16 v16h;
typedef __attribute__((ext_vector_type(8)))  _Float16 v8h;
typedef __attribute__((ext_vector_type(8)))  float    v8f;
typedef __attribute__((ext_vector_type(4)))  unsigned int u32x4;

#define N_ATOMS   100000
#define N_BONDS   200000
#define MAX_NB    6
#define ATOM_FDIM 133
#define BOND_FDIM 147
#define HIDDEN    300
#define N_MOL     5000

// padded GEMM dims (K multiple of 32, N multiple of 16)
#define KI 160   // 147 -> 160   (f_bonds @ W_i)
#define KH 320   // 300 -> 320   (message @ W_h)
#define KO 448   // 433 -> 448   (a_input @ W_o)
#define NP 304   // 300 -> 304
#define NT (NP/16)            // 19 N-tiles
#define MT_B (N_BONDS/16)     // 12500 (exact)
#define MT_A (N_ATOMS/16)     // 6250  (exact)

static __host__ __device__ inline int cdiv(int a, int b) { return (a + b - 1) / b; }

// ---------------------------------------------------------------------------
// WMMA core: one wave computes a 16x16 f32 tile, K loop in steps of 32.
// A layout: lane l: row = l&15, 16 contiguous K values starting at
// (l>>4)*16. B (from LDS, pre-transposed weights):
// lane l: col = l&15, 16 contiguous K values starting at (l>>4)*16.
// ---------------------------------------------------------------------------
template <int KPAD>
__device__ __forceinline__ v8f wmma_mac(const _Float16* __restrict__ arow,
                                        const _Float16* bl, int ahi) {
  v8f acc = {0.f, 0.f, 0.f, 0.f, 0.f, 0.f, 0.f, 0.f};
#pragma unroll
  for (int ks = 0; ks < KPAD; ks += 32) {
    v8h a0 = *(const v8h*)(arow + ks + 2 * ahi);
    v8h a1 = *(const v8h*)(arow + ks + 2 * ahi + 8);
    v16h a = __builtin_shufflevector(a0, a1, 0, 1, 2, 3, 4, 5, 6, 7,
                                     8, 9, 10, 11, 12, 13, 14, 15);
    v16h b = *(const v16h*)(bl + ks);
    acc = __builtin_amdgcn_wmma_f32_16x16x32_f16(
        /*neg_a=*/false, a, /*neg_b=*/false, b,
        /*c_mod=*/(short)0, acc, /*reuse_a=*/false, /*reuse_b=*/false);
  }
  return acc;
}

// cooperative load of one 16xKPAD B tile (transposed weights) into LDS
template <int KPAD>
__device__ __forceinline__ void load_btile(_Float16* Bl,
                                           const _Float16* __restrict__ Bt,
                                           int nbase) {
  const int chunks = KPAD / 8;
  for (int i = threadIdx.x; i < 16 * chunks; i += blockDim.x) {
    int nl = i / chunks, ck = i % chunks;
    *(v8h*)(Bl + nl * KPAD + ck * 8) =
        *(const v8h*)(Bt + (size_t)(nbase + nl) * KPAD + ck * 8);
  }
  __syncthreads();
}

// ---------------------------------------------------------------------------
// GEMM 1: inp = f_bonds16 @ W_i^T ; message = relu(inp)
// ---------------------------------------------------------------------------
__global__ void __launch_bounds__(256)
k_gemm_inp(const _Float16* __restrict__ A, const _Float16* __restrict__ Bt,
           _Float16* __restrict__ inp16, _Float16* __restrict__ msg16) {
  __shared__ __attribute__((aligned(32))) _Float16 Bl[16 * KI];
  int nt = blockIdx.x % NT, mtg = blockIdx.x / NT;
  int nbase = nt * 16;
  load_btile<KI>(Bl, Bt, nbase);
  int mt = mtg * 8 + (threadIdx.x >> 5);
  if (mt >= MT_B) return;
  int lane = threadIdx.x & 31, lo = lane & 15, hi = lane >> 4;
  const _Float16* arow = A + (size_t)(mt * 16 + lo) * KI;
  v8f acc = wmma_mac<KI>(arow, &Bl[lo * KI + hi * 16], hi * 8);
  int n = nbase + lo;
  if (n < HIDDEN) {
    int mrow = mt * 16 + hi * 8;
#pragma unroll
    for (int r = 0; r < 8; ++r) {
      float v = acc[r];
      inp16[(size_t)(mrow + r) * NP + n] = (_Float16)v;
      msg16[(size_t)(mrow + r) * KH + n] = (_Float16)fmaxf(v, 0.f);
    }
  }
}

// ---------------------------------------------------------------------------
// GEMM 2: message = relu(inp + mpre @ W_h^T)
// ---------------------------------------------------------------------------
__global__ void __launch_bounds__(256)
k_gemm_h(const _Float16* __restrict__ A, const _Float16* __restrict__ Bt,
         const _Float16* __restrict__ inp16, _Float16* __restrict__ msg16) {
  __shared__ __attribute__((aligned(32))) _Float16 Bl[16 * KH];
  int nt = blockIdx.x % NT, mtg = blockIdx.x / NT;
  int nbase = nt * 16;
  load_btile<KH>(Bl, Bt, nbase);
  int mt = mtg * 8 + (threadIdx.x >> 5);
  if (mt >= MT_B) return;
  int lane = threadIdx.x & 31, lo = lane & 15, hi = lane >> 4;
  const _Float16* arow = A + (size_t)(mt * 16 + lo) * KH;
  v8f acc = wmma_mac<KH>(arow, &Bl[lo * KH + hi * 16], hi * 8);
  int n = nbase + lo;
  if (n < HIDDEN) {
    int mrow = mt * 16 + hi * 8;
#pragma unroll
    for (int r = 0; r < 8; ++r) {
      float v = (float)inp16[(size_t)(mrow + r) * NP + n] + acc[r];
      msg16[(size_t)(mrow + r) * KH + n] = (_Float16)fmaxf(v, 0.f);
    }
  }
}

// ---------------------------------------------------------------------------
// GEMM 3: atom_hiddens = relu(a_input @ W_o^T + b_o)   (f32 out)
// ---------------------------------------------------------------------------
__global__ void __launch_bounds__(256)
k_gemm_o(const _Float16* __restrict__ A, const _Float16* __restrict__ Bt,
         const float* __restrict__ b_o, float* __restrict__ hid) {
  __shared__ __attribute__((aligned(32))) _Float16 Bl[16 * KO];
  int nt = blockIdx.x % NT, mtg = blockIdx.x / NT;
  int nbase = nt * 16;
  load_btile<KO>(Bl, Bt, nbase);
  int mt = mtg * 8 + (threadIdx.x >> 5);
  if (mt >= MT_A) return;
  int lane = threadIdx.x & 31, lo = lane & 15, hi = lane >> 4;
  const _Float16* arow = A + (size_t)(mt * 16 + lo) * KO;
  v8f acc = wmma_mac<KO>(arow, &Bl[lo * KO + hi * 16], hi * 8);
  int n = nbase + lo;
  if (n < HIDDEN) {
    float bias = b_o[n];
    int mrow = mt * 16 + hi * 8;
#pragma unroll
    for (int r = 0; r < 8; ++r)
      hid[(size_t)(mrow + r) * NP + n] = fmaxf(acc[r] + bias, 0.f);
  }
}

// ---------------------------------------------------------------------------
// elementwise / gather kernels
// ---------------------------------------------------------------------------
__global__ void k_zero16(u32x4* p, int n4) {
  int i = blockIdx.x * blockDim.x + threadIdx.x;
  if (i < n4) p[i] = (u32x4){0u, 0u, 0u, 0u};
}

__global__ void k_prep_fb(const float* __restrict__ fb, _Float16* __restrict__ fb16) {
  int i = blockIdx.x * blockDim.x + threadIdx.x;
  if (i >= N_BONDS * KI) return;
  int b = i / KI, k = i % KI;
  fb16[i] = (k < BOND_FDIM) ? (_Float16)fb[(size_t)b * BOND_FDIM + k] : (_Float16)0.f;
}

// transpose+pad weights: src [K, HIDDEN] row-major -> dst [NP, KPAD] f16
__global__ void k_prep_wt(const float* __restrict__ src, _Float16* __restrict__ dst,
                          int K, int KPADv) {
  int i = blockIdx.x * blockDim.x + threadIdx.x;
  if (i >= NP * KPADv) return;
  int n = i / KPADv, k = i % KPADv;
  float v = (n < HIDDEN && k < K) ? src[(size_t)k * HIDDEN + n] : 0.f;
  dst[i] = (_Float16)v;
}

// a_message[a] = sum_j message[a2b[a][j]]   (8 cols / thread, f32 accum)
__global__ void k_gather(const _Float16* __restrict__ msg, const int* __restrict__ a2b,
                         _Float16* __restrict__ amsg) {
  int i = blockIdx.x * blockDim.x + threadIdx.x;
  if (i >= N_ATOMS * (KH / 8)) return;
  int a = i / (KH / 8), c = i % (KH / 8);
  float s[8] = {0, 0, 0, 0, 0, 0, 0, 0};
#pragma unroll
  for (int j = 0; j < MAX_NB; ++j) {
    int b = a2b[a * MAX_NB + j];
    v8h v = *(const v8h*)(msg + (size_t)b * KH + c * 8);
#pragma unroll
    for (int e = 0; e < 8; ++e) s[e] += (float)v[e];
  }
  v8h o;
#pragma unroll
  for (int e = 0; e < 8; ++e) o[e] = (_Float16)s[e];
  *(v8h*)(amsg + (size_t)a * KH + c * 8) = o;
}

// mpre[b] = a_message[b2a[b]] - message[b2revb[b]]
__global__ void k_mpre(const _Float16* __restrict__ amsg, const _Float16* __restrict__ msg,
                       const int* __restrict__ b2a, const int* __restrict__ b2revb,
                       _Float16* __restrict__ mpre) {
  int i = blockIdx.x * blockDim.x + threadIdx.x;
  if (i >= N_BONDS * (KH / 8)) return;
  int b = i / (KH / 8), c = i % (KH / 8);
  int sa = b2a[b], rb = b2revb[b];
  v8h x = *(const v8h*)(amsg + (size_t)sa * KH + c * 8);
  v8h y = *(const v8h*)(msg + (size_t)rb * KH + c * 8);
  v8h o;
#pragma unroll
  for (int e = 0; e < 8; ++e) o[e] = (_Float16)((float)x[e] - (float)y[e]);
  *(v8h*)(mpre + (size_t)b * KH + c * 8) = o;
}

// a_input = concat(f_atoms, a_message), zero-padded to KO
__global__ void k_ainput(const float* __restrict__ fa, const _Float16* __restrict__ amsg,
                         _Float16* __restrict__ ain) {
  int i = blockIdx.x * blockDim.x + threadIdx.x;
  if (i >= N_ATOMS * KO) return;
  int a = i / KO, c = i % KO;
  float v;
  if (c < ATOM_FDIM)                 v = fa[(size_t)a * ATOM_FDIM + c];
  else if (c < ATOM_FDIM + HIDDEN)   v = (float)amsg[(size_t)a * KH + (c - ATOM_FDIM)];
  else                               v = 0.f;
  ain[i] = (_Float16)v;
}

// per-molecule mean via binary search on sorted atom_mol_ids (deterministic)
__device__ __forceinline__ int lower_bound_i(const int* a, int n, int key) {
  int lo = 0, hi = n;
  while (lo < hi) { int mid = (lo + hi) >> 1; if (a[mid] < key) lo = mid + 1; else hi = mid; }
  return lo;
}

__global__ void __launch_bounds__(320)
k_molmean(const float* __restrict__ hid, const int* __restrict__ ids,
          float* __restrict__ out) {
  int mol = blockIdx.x;
  int s = lower_bound_i(ids, N_ATOMS, mol);
  int e = lower_bound_i(ids, N_ATOMS, mol + 1);
  int k = threadIdx.x;
  if (k >= HIDDEN) return;
  float acc = 0.f;
  for (int a = s; a < e; ++a) acc += hid[(size_t)a * NP + k];
  int cnt = e - s;
  out[(size_t)mol * HIDDEN + k] = cnt > 0 ? acc / (float)cnt : 0.f;
}

// ---------------------------------------------------------------------------
extern "C" void kernel_launch(void* const* d_in, const int* in_sizes, int n_in,
                              void* d_out, int out_size, void* d_ws, size_t ws_size,
                              hipStream_t stream) {
  const float* f_atoms = (const float*)d_in[0];
  const float* f_bonds = (const float*)d_in[1];
  const float* W_i     = (const float*)d_in[2];
  const float* W_h     = (const float*)d_in[3];
  const float* W_o     = (const float*)d_in[4];
  const float* b_o     = (const float*)d_in[5];
  const int*   a2b     = (const int*)d_in[6];
  const int*   b2a     = (const int*)d_in[7];
  const int*   b2revb  = (const int*)d_in[8];
  const int*   mol_ids = (const int*)d_in[9];
  float* out = (float*)d_out;

  // workspace layout (bytes, 256-aligned regions), ~442 MB total
  char* w = (char*)d_ws;
  _Float16* inp16  = (_Float16*)(w + 0);              // 200000*304*2 = 121,600,000
  _Float16* msg16  = (_Float16*)(w + 121600000);      // 200000*320*2 = 128,000,000
  _Float16* mpre16 = (_Float16*)(w + 249600000);      // 128,000,000 (aliased below)
  _Float16* amsg16 = (_Float16*)(w + 377600000);      // 100000*320*2 = 64,000,000
  _Float16* WtI    = (_Float16*)(w + 441600000);      // 304*160*2
  _Float16* WtH    = (_Float16*)(w + 441697280);      // 304*320*2
  _Float16* WtO    = (_Float16*)(w + 441891840);      // 304*448*2  (end 442,164,224)
  _Float16* fb16   = mpre16;                          // alias (phase 1 only)
  _Float16* ain16  = mpre16;                          // alias (after depth loop)
  float*    hid    = (float*)inp16;                   // alias (after last gemm_h)

  const int TPB = 256;

  // 0) zero msg16 so K-pad columns [300,320) stay 0 forever
  {
    int n4 = 200000 * 320 * 2 / 16;
    k_zero16<<<cdiv(n4, TPB), TPB, 0, stream>>>((u32x4*)msg16, n4);
  }
  // 1) weight / input prep
  k_prep_fb<<<cdiv(N_BONDS * KI, TPB), TPB, 0, stream>>>(f_bonds, fb16);
  k_prep_wt<<<cdiv(NP * KI, TPB), TPB, 0, stream>>>(W_i, WtI, BOND_FDIM, KI);
  k_prep_wt<<<cdiv(NP * KH, TPB), TPB, 0, stream>>>(W_h, WtH, HIDDEN, KH);
  k_prep_wt<<<cdiv(NP * KO, TPB), TPB, 0, stream>>>(W_o, WtO, ATOM_FDIM + HIDDEN, KO);

  // 2) inp = f_bonds @ W_i ; message = relu(inp)
  {
    dim3 grid(NT * cdiv(MT_B, 8));
    k_gemm_inp<<<grid, TPB, 0, stream>>>(fb16, WtI, inp16, msg16);
  }

  // 3) DEPTH-1 = 2 message-passing steps
  for (int it = 0; it < 2; ++it) {
    k_gather<<<cdiv(N_ATOMS * (KH / 8), TPB), TPB, 0, stream>>>(msg16, a2b, amsg16);
    k_mpre<<<cdiv(N_BONDS * (KH / 8), TPB), TPB, 0, stream>>>(amsg16, msg16, b2a, b2revb, mpre16);
    dim3 grid(NT * cdiv(MT_B, 8));
    k_gemm_h<<<grid, TPB, 0, stream>>>(mpre16, WtH, inp16, msg16);
  }

  // 4) final gather + readout GEMM
  k_gather<<<cdiv(N_ATOMS * (KH / 8), TPB), TPB, 0, stream>>>(msg16, a2b, amsg16);
  k_ainput<<<cdiv(N_ATOMS * KO, TPB), TPB, 0, stream>>>(f_atoms, amsg16, ain16);
  {
    dim3 grid(NT * cdiv(MT_A, 8));
    k_gemm_o<<<grid, TPB, 0, stream>>>(ain16, WtO, b_o, hid);
  }

  // 5) per-molecule mean (deterministic, no atomics)
  k_molmean<<<N_MOL, 320, 0, stream>>>(hid, mol_ids, out);
}
